// QuantumNeuralLayer_6957847020213
// MI455X (gfx1250) — matrix operand order repaired and checked by reading, not compiled, not run on hardware
//
#include <hip/hip_runtime.h>
#include <hip/hip_bf16.h>

// ---------------------------------------------------------------------------
// QuantumNeuralLayer for MI455X (gfx1250, wave32, WMMA)
//
//  x(2048x4096 f32) -> 2x [relu(GEMM)] -> row-normalize -> 12-qubit circuit
//  (4 layers x 12 qubits x RxRyRz + CNOT chains) -> real() @ W_int ->
//  2x [relu(GEMM)] -> out f32
//
//  * 5 GEMMs (2048x4096x4096) on bf16 WMMA, f32 accumulate.
//  * A-tiles staged with GLOBAL_LOAD_ASYNC_TO_LDS_B128 (ASYNCcnt path),
//    double-buffered LDS tiles overlap staging with WMMA.
//  * Quantum state (32KB/row) lives entirely in LDS; one HBM read+write of
//    the 64MB state instead of ~18GB for a naive global-memory circuit.
// ---------------------------------------------------------------------------

typedef __attribute__((ext_vector_type(16))) __bf16 v16bf;
typedef __attribute__((ext_vector_type(8)))  __bf16 bf16x8;
typedef __attribute__((ext_vector_type(8)))  float  v8f;

#define TM 128
#define TN 128
#define TK 32
#define LDT 40                 // padded LDS row stride (80B) -> conflict-lean b128
#define A_ELEMS (TM * LDT)     // 5120 bf16 per A buffer
#define B_ELEMS (TN * LDT)     // 5120 bf16 per B buffer

// ---------------- f32 -> bf16 convert (prologue) ----------------
__global__ __launch_bounds__(256)
void cvt_f32_bf16(const float* __restrict__ in, __bf16* __restrict__ out, int n) {
    int i = (blockIdx.x * 256 + threadIdx.x) * 4;
    if (i + 3 < n) {
        float4 v = *(const float4*)(in + i);
        out[i + 0] = (__bf16)v.x;
        out[i + 1] = (__bf16)v.y;
        out[i + 2] = (__bf16)v.z;
        out[i + 3] = (__bf16)v.w;
    }
}

// ---------------- bf16 WMMA GEMM: C = act( A @ op(W) + bias ) ----------------
// A:  M x K bf16 (row major activations)
// W:  f32 weights. transB=1: B[k][n] = W[n*K+k]  (x @ Wcls.T)
//                  transB=0: B[k][n] = W[k*N+n]  (feat @ W_int)
__global__ __launch_bounds__(256)
void gemm_bf16_wmma(const __bf16* __restrict__ A, const float* __restrict__ W,
                    const float* __restrict__ bias, __bf16* __restrict__ outBf,
                    float* __restrict__ outF32, int M, int N, int K,
                    int transB, int doRelu)
{
    // Single shared array => its base LDS offset is 0, so async-to-LDS byte
    // addresses are just element_index * sizeof(bf16).
    __shared__ __bf16 smem[2 * A_ELEMS + 2 * B_ELEMS];   // 40 KB

    const int t    = threadIdx.x;
    const int lane = t & 31;
    const int w    = t >> 5;      // 8 waves
    const int wm   = w & 1;       // 2 wave-rows  -> 64 rows each
    const int wn   = w >> 1;      // 4 wave-cols  -> 32 cols each
    const int mBase = blockIdx.y * TM;
    const int nBase = blockIdx.x * TN;

    const int r = lane & 15;      // fragment row/col within 16
    const int h = lane >> 4;      // lane half selects K sub-range

    v8f acc[4][2];
    #pragma unroll
    for (int i = 0; i < 4; ++i)
        #pragma unroll
        for (int j = 0; j < 2; ++j) {
            v8f z = {};
            acc[i][j] = z;
        }

    union FragU { v16bf v; bf16x8 p[2]; };

    // ---- tile stagers --------------------------------------------------
    // A tile: 128x32 bf16 copied global->LDS via async DMA (no VGPR bounce)
    auto stageA = [&](int buf, int kb) {
        #pragma unroll
        for (int i = 0; i < 2; ++i) {
            int c   = t + 256 * i;          // 0..511 chunks of 8 bf16
            int row = c >> 2;
            int col = (c & 3) * 8;
            unsigned lds = (unsigned)((buf * A_ELEMS + row * LDT + col) * 2);
            const __bf16* g = A + (size_t)(mBase + row) * K + kb + col;
            asm volatile("global_load_async_to_lds_b128 %0, %1, off"
                         :: "v"(lds), "v"(g) : "memory");
        }
    };
    // B tile: f32 weights -> bf16 in regs -> LDS as [n][k]
    auto stageB = [&](int buf, int kb) {
        __bf16* sB = &smem[2 * A_ELEMS + buf * B_ELEMS];
        if (transB) {
            #pragma unroll
            for (int i = 0; i < 2; ++i) {
                int c   = t + 256 * i;      // 0..511
                int n   = c >> 2;
                int col = (c & 3) * 8;
                const float* src = W + (size_t)(nBase + n) * K + kb + col;
                float4 f0 = *(const float4*)(src);
                float4 f1 = *(const float4*)(src + 4);
                bf16x8 p;
                p[0] = (__bf16)f0.x; p[1] = (__bf16)f0.y;
                p[2] = (__bf16)f0.z; p[3] = (__bf16)f0.w;
                p[4] = (__bf16)f1.x; p[5] = (__bf16)f1.y;
                p[6] = (__bf16)f1.z; p[7] = (__bf16)f1.w;
                *(bf16x8*)(&sB[n * LDT + col]) = p;
            }
        } else {
            #pragma unroll
            for (int i = 0; i < 4; ++i) {
                int c  = t + 256 * i;       // 0..1023
                int k  = c >> 5;            // 0..31
                int n4 = (c & 31) * 4;      // 0..124
                float4 f = *(const float4*)(W + (size_t)(kb + k) * N + nBase + n4);
                sB[(n4 + 0) * LDT + k] = (__bf16)f.x;
                sB[(n4 + 1) * LDT + k] = (__bf16)f.y;
                sB[(n4 + 2) * LDT + k] = (__bf16)f.z;
                sB[(n4 + 3) * LDT + k] = (__bf16)f.w;
            }
        }
    };

    // ---- software-pipelined main loop ----------------------------------
    stageA(0, 0);
    stageB(0, 0);
    asm volatile("s_wait_asynccnt 0" ::: "memory");
    __syncthreads();

    const int nk = K / TK;
    for (int kt = 0; kt < nk; ++kt) {
        const int cur = kt & 1;
        const int kb  = kt * TK;
        if (kt + 1 < nk) {              // prefetch next tiles into alt buffers
            stageA(cur ^ 1, kb + TK);
            stageB(cur ^ 1, kb + TK);
        }

        const __bf16* sA = &smem[cur * A_ELEMS];
        const __bf16* sB = &smem[2 * A_ELEMS + cur * B_ELEMS];

        // B fragments: lane holds N=r, K = {h*8..h*8+7, h*8+16..h*8+23}
        FragU bf[2];
        #pragma unroll
        for (int ni = 0; ni < 2; ++ni) {
            const __bf16* base = &sB[(wn * 32 + ni * 16 + r) * LDT + h * 8];
            bf[ni].p[0] = *(const bf16x8*)(base);
            bf[ni].p[1] = *(const bf16x8*)(base + 16);
        }
        // A fragments: lane holds M=r, K = {h*8..h*8+7, h*8+16..h*8+23}
        #pragma unroll
        for (int mi = 0; mi < 4; ++mi) {
            FragU af;
            const __bf16* base = &sA[(wm * 64 + mi * 16 + r) * LDT + h * 8];
            af.p[0] = *(const bf16x8*)(base);
            af.p[1] = *(const bf16x8*)(base + 16);
            #pragma unroll
            for (int ni = 0; ni < 2; ++ni) {
                acc[mi][ni] = __builtin_amdgcn_wmma_f32_16x16x32_bf16(
                    false, af.v, false, bf[ni].v, (short)0, acc[mi][ni],
                    false, false);
            }
        }

        // next-tile async copies must land; __syncthreads only fences
        // LOAD/STORE/DS counters, ASYNCcnt is waited explicitly.
        asm volatile("s_wait_asynccnt 0" ::: "memory");
        __syncthreads();
    }

    // ---- epilogue: D lane layout -> M = e + 8*h, N = r ----
    #pragma unroll
    for (int mi = 0; mi < 4; ++mi) {
        #pragma unroll
        for (int ni = 0; ni < 2; ++ni) {
            int n = nBase + wn * 32 + ni * 16 + r;
            float bv = bias ? bias[n] : 0.0f;
            #pragma unroll
            for (int e = 0; e < 8; ++e) {
                int m = mBase + wm * 64 + mi * 16 + h * 8 + e;
                float v = acc[mi][ni][e] + bv;
                if (doRelu) v = fmaxf(v, 0.0f);
                if (outF32) outF32[(size_t)m * N + n] = v;
                if (outBf)  outBf [(size_t)m * N + n] = (__bf16)v;
            }
        }
    }
}

// ---------------- quantum circuit, state in LDS ----------------
// One workgroup per batch row. re/im f32 state (32KB) + 48 fused U matrices.
__global__ __launch_bounds__(256)
void quantum_circuit(const __bf16* __restrict__ fin, const float* __restrict__ qp,
                     __bf16* __restrict__ outBf)
{
    __shared__ float re[4096];
    __shared__ float im[4096];
    __shared__ float umat[48 * 8];
    __shared__ float red[256];

    const int t = threadIdx.x;
    const size_t rowOff = (size_t)blockIdx.x * 4096;

    // fused U = Rz(a2) @ Ry(a1) @ Rx(a0), one gate per thread (48 gates)
    if (t < 48) {
        float a0 = qp[t * 3 + 0] * 0.5f;
        float a1 = qp[t * 3 + 1] * 0.5f;
        float a2 = qp[t * 3 + 2] * 0.5f;
        float c0 = cosf(a0), s0 = sinf(a0);
        float c1 = cosf(a1), s1 = sinf(a1);
        float cz = cosf(a2), sz = sinf(a2);
        // M = Ry@Rx
        float m00r =  c1 * c0, m00i =  s1 * s0;
        float m01r = -s1 * c0, m01i = -c1 * s0;
        float m10r =  s1 * c0, m10i = -c1 * s0;
        float m11r =  c1 * c0, m11i = -s1 * s0;
        // row0 *= e^{-i a2/2}, row1 *= e^{+i a2/2}
        umat[t * 8 + 0] = cz * m00r + sz * m00i;
        umat[t * 8 + 1] = cz * m00i - sz * m00r;
        umat[t * 8 + 2] = cz * m01r + sz * m01i;
        umat[t * 8 + 3] = cz * m01i - sz * m01r;
        umat[t * 8 + 4] = cz * m10r - sz * m10i;
        umat[t * 8 + 5] = cz * m10i + sz * m10r;
        umat[t * 8 + 6] = cz * m11r - sz * m11i;
        umat[t * 8 + 7] = cz * m11i + sz * m11r;
    }

    // load row, row-norm
    float s = 0.0f;
    #pragma unroll
    for (int j = 0; j < 16; ++j) {
        int i = t * 16 + j;
        float v = (float)fin[rowOff + i];
        re[i] = v; im[i] = 0.0f;
        s += v * v;
    }
    red[t] = s;
    __syncthreads();
    for (int st = 128; st > 0; st >>= 1) {
        if (t < st) red[t] += red[t + st];
        __syncthreads();
    }
    float invn = 1.0f / fmaxf(sqrtf(red[0]), 1e-12f);
    #pragma unroll
    for (int j = 0; j < 16; ++j) re[t * 16 + j] *= invn;

    // circuit: both elements of each amplitude pair owned by one thread
    for (int l = 0; l < 4; ++l) {
        for (int q = 0; q < 12; ++q) {
            __syncthreads();
            const float* u = &umat[(l * 12 + q) * 8];
            float u00r = u[0], u00i = u[1], u01r = u[2], u01i = u[3];
            float u10r = u[4], u10i = u[5], u11r = u[6], u11i = u[7];
            int low = 1 << q;
            #pragma unroll
            for (int j = 0; j < 8; ++j) {
                int p  = t + 256 * j;
                int i0 = ((p >> q) << (q + 1)) | (p & (low - 1));
                int i1 = i0 | low;
                float ar = re[i0], ai = im[i0];
                float br = re[i1], bi = im[i1];
                re[i0] = u00r * ar - u00i * ai + u01r * br - u01i * bi;
                im[i0] = u00r * ai + u00i * ar + u01r * bi + u01i * br;
                re[i1] = u10r * ar - u10i * ai + u11r * br - u11i * bi;
                im[i1] = u10r * ai + u10i * ar + u11r * bi + u11i * br;
            }
        }
        if (l < 3) {
            // fused even+odd CNOT chains: src = EVEN(ODD(idx)); disjoint
            // control/target bit pairs commute -> pure bit arithmetic.
            __syncthreads();
            float tr[16], ti[16];
            #pragma unroll
            for (int j = 0; j < 16; ++j) {
                int idx = t * 16 + j;
                int so  = idx ^ ((idx & 0x2AA) << 1);
                int src = so  ^ ((so  & 0x555) << 1);
                tr[j] = re[src]; ti[j] = im[src];
            }
            __syncthreads();
            #pragma unroll
            for (int j = 0; j < 16; ++j) {
                re[t * 16 + j] = tr[j];
                im[t * 16 + j] = ti[j];
            }
        }
    }

    __syncthreads();
    #pragma unroll
    for (int j = 0; j < 16; ++j) {
        int i = t * 16 + j;
        outBf[rowOff + i] = (__bf16)re[i];
    }
}

// ---------------- driver ----------------
extern "C" void kernel_launch(void* const* d_in, const int* in_sizes, int n_in,
                              void* d_out, int out_size, void* d_ws, size_t ws_size,
                              hipStream_t stream)
{
    const float* x    = (const float*)d_in[0];   // 2048 x 4096
    const float* qp   = (const float*)d_in[1];   // 4 x 12 x 3
    const float* Wcls = (const float*)d_in[2];   // 4 x 4096 x 4096
    const float* bcls = (const float*)d_in[3];   // 4 x 4096
    const float* Wint = (const float*)d_in[4];   // 4096 x 4096
    float* out = (float*)d_out;

    const int M = 2048, D = 4096;
    const size_t MD = (size_t)M * D;

    __bf16* actA = (__bf16*)d_ws;      // 16 MB
    __bf16* actB = actA + MD;          // 16 MB

    dim3 blk(256);
    dim3 gcv((unsigned)(MD / (256 * 4)));
    dim3 gg(D / TN, M / TM);

    cvt_f32_bf16<<<gcv, blk, 0, stream>>>(x, actA, (int)MD);

    // classical front half: relu(f @ Wcls[l].T + b[l])
    gemm_bf16_wmma<<<gg, blk, 0, stream>>>(actA, Wcls + 0 * (size_t)D * D,
                                           bcls + 0 * D, actB, nullptr,
                                           M, D, D, 1, 1);
    gemm_bf16_wmma<<<gg, blk, 0, stream>>>(actB, Wcls + 1 * (size_t)D * D,
                                           bcls + 1 * D, actA, nullptr,
                                           M, D, D, 1, 1);

    // normalize + 12-qubit circuit, state in LDS
    quantum_circuit<<<dim3(M), blk, 0, stream>>>(actA, qp, actB);

    // feat = real(state) @ W_int   (no bias, no relu)
    gemm_bf16_wmma<<<gg, blk, 0, stream>>>(actB, Wint, nullptr, actA, nullptr,
                                           M, D, D, 0, 0);

    // classical back half
    gemm_bf16_wmma<<<gg, blk, 0, stream>>>(actA, Wcls + 2 * (size_t)D * D,
                                           bcls + 2 * D, actB, nullptr,
                                           M, D, D, 1, 1);
    gemm_bf16_wmma<<<gg, blk, 0, stream>>>(actB, Wcls + 3 * (size_t)D * D,
                                           bcls + 3 * D, nullptr, out,
                                           M, D, D, 1, 1);
}
